// GINEncoder_35519379538034
// MI455X (gfx1250) — matrix operand order repaired and checked by reading, not compiled, not run on hardware
//
#include <hip/hip_runtime.h>
#include <hip/hip_bf16.h>

// ---------------------------------------------------------------------------
// GIN encoder for MI455X (gfx1250, wave32).
//   - edge scatter: 1 wave / edge, float4 gather + global_atomic_add_f32
//   - fused GEMMs:  v_wmma_f32_16x16x32_f16, 16x64 tile per wave,
//                   weights pre-swizzled in LDS into fragment layout so each
//                   B fragment is two ds_load_b128 per lane
//   - batchnorm:    two-pass (atomic column sums, then normalize+relu)
//   - pooling:      atomic segment-sum into d_out, then divide by counts
// ---------------------------------------------------------------------------

typedef __attribute__((ext_vector_type(16))) _Float16 v16h;
typedef __attribute__((ext_vector_type(8)))  float    v8f;

#define BN_EPS 1e-5f

// ---------------------------- edge aggregation -----------------------------

__global__ __launch_bounds__(256) void k_agg128(
    const float* __restrict__ x, const int* __restrict__ src,
    const int* __restrict__ dst, float* __restrict__ agg, int nedges)
{
  int w    = (int)((blockIdx.x * 256u + threadIdx.x) >> 5);
  int lane = threadIdx.x & 31;
  if (w >= nedges) return;
  int s = src[w], d = dst[w];
  const float4 v = ((const float4*)(x + (size_t)s * 128))[lane];   // 512B/wave
  float* o = agg + (size_t)d * 128 + lane * 4;
  atomicAdd(o + 0, v.x); atomicAdd(o + 1, v.y);
  atomicAdd(o + 2, v.z); atomicAdd(o + 3, v.w);
}

__global__ __launch_bounds__(256) void k_agg64(
    const float* __restrict__ h, const int* __restrict__ src,
    const int* __restrict__ dst, float* __restrict__ agg, int nedges)
{
  int w    = (int)((blockIdx.x * 256u + threadIdx.x) >> 5);
  int lane = threadIdx.x & 31;
  if (w >= nedges) return;
  int s = src[w], d = dst[w];
  const float2 v = ((const float2*)(h + (size_t)s * 64))[lane];
  float* o = agg + (size_t)d * 64 + lane * 2;
  atomicAdd(o + 0, v.x); atomicAdd(o + 1, v.y);
}

// ------------------------------ fused GEMM ---------------------------------
// out[M x 64] = act( (in1 (+ in2)) [M x K] @ W [K x 64] + bias ), f16 WMMA,
// fp32 accumulate. Block = 256 threads = 8 waves; wave owns a 16x64 tile.
//
// LDS weight layout is pre-swizzled into WMMA B-fragment order:
//   frag f = (kchunk*4 + ntile), element [lane][e] (16 halves = 32B per lane)
//   maps to W[kchunk*32 + (e>>3)*16 + (lane>>4)*8 + (e&7)][ntile*16 + (lane&15)]
// so the inner loop reads one aligned v16h (2x ds_load_b128) per fragment.

template<int K, bool ADD2, bool RELU>
__global__ __launch_bounds__(256) void k_gemm(
    const float* __restrict__ in1, const float* __restrict__ in2,
    const float* __restrict__ W,  const float* __restrict__ bias,
    float* __restrict__ out, int nrows)
{
  __shared__ __align__(16) _Float16 ldsW[K * 64];
  __shared__ float ldsBias[64];

  for (int idx = threadIdx.x; idx < K * 64; idx += 256) {
    const int e    = idx & 15;
    const int ln   = (idx >> 4) & 31;
    const int nt   = (idx >> 9) & 3;
    const int kc   = idx >> 11;
    const int krow = kc * 32 + (e & 8) * 2 + (ln >> 4) * 8 + (e & 7);
    const int col  = nt * 16 + (ln & 15);
    ldsW[idx] = (_Float16)W[krow * 64 + col];
  }
  if (threadIdx.x < 64) ldsBias[threadIdx.x] = bias[threadIdx.x];
  __syncthreads();

  const int lane = threadIdx.x & 31;
  const int wave = threadIdx.x >> 5;
  const int row0 = (blockIdx.x * 8 + wave) * 16;
  if (row0 >= nrows) return;                 // wave-uniform: EXEC stays all-ones

  const int m  = lane & 15;
  const int hi = lane >> 4;
  int row = row0 + m; if (row >= nrows) row = nrows - 1;   // clamp (tail tiles)

  const v16h* wfrag = (const v16h*)ldsW;
  v8f acc[4] = {};

#pragma unroll
  for (int kc = 0; kc < K; kc += 32) {
    // ---- A fragment (16x32 f16): lanes 0-15 hold K kb+[0..7] and kb+[16..23]
    const int kb = kc + (hi ? 8 : 0);        // multiple of 8 -> 16B aligned
    const float4* q1 = (const float4*)(in1 + (size_t)row * K + kb);
    float4 a0 = q1[0], a1 = q1[1];           // kb   .. kb+7
    float4 a2 = q1[4], a3 = q1[5];           // kb+16.. kb+23
    if (ADD2) {
      const float4* q2 = (const float4*)(in2 + (size_t)row * K + kb);
      float4 c0 = q2[0], c1 = q2[1], c2 = q2[4], c3 = q2[5];
      a0.x += c0.x; a0.y += c0.y; a0.z += c0.z; a0.w += c0.w;
      a1.x += c1.x; a1.y += c1.y; a1.z += c1.z; a1.w += c1.w;
      a2.x += c2.x; a2.y += c2.y; a2.z += c2.z; a2.w += c2.w;
      a3.x += c3.x; a3.y += c3.y; a3.z += c3.z; a3.w += c3.w;
    }
    v16h a;
    a[0]  = (_Float16)a0.x; a[1]  = (_Float16)a0.y;
    a[2]  = (_Float16)a0.z; a[3]  = (_Float16)a0.w;
    a[4]  = (_Float16)a1.x; a[5]  = (_Float16)a1.y;
    a[6]  = (_Float16)a1.z; a[7]  = (_Float16)a1.w;
    a[8]  = (_Float16)a2.x; a[9]  = (_Float16)a2.y;
    a[10] = (_Float16)a2.z; a[11] = (_Float16)a2.w;
    a[12] = (_Float16)a3.x; a[13] = (_Float16)a3.y;
    a[14] = (_Float16)a3.z; a[15] = (_Float16)a3.w;

    // ---- 4 B fragments, each one aligned 32B LDS read per lane
    const int fbase = (kc >> 5) * 4;
#pragma unroll
    for (int nt = 0; nt < 4; ++nt) {
      v16h b = wfrag[(fbase + nt) * 32 + lane];
      acc[nt] = __builtin_amdgcn_wmma_f32_16x16x32_f16(
          false, a, false, b, (short)0, acc[nt], false, false);
    }
  }

  // ---- epilogue; D layout: VGPR r -> row (row0 + r + 8*hi), col = nt*16 + m
  if (row0 + 16 <= nrows) {                  // full tile: unconditional stores
#pragma unroll
    for (int nt = 0; nt < 4; ++nt) {
      const int col = nt * 16 + m;
      const float bs = ldsBias[col];
#pragma unroll
      for (int r = 0; r < 8; ++r) {
        const int rr = row0 + r + (hi ? 8 : 0);
        float v = acc[nt][r] + bs;
        if (RELU) v = fmaxf(v, 0.0f);
        out[(size_t)rr * 64 + col] = v;
      }
    }
  } else {                                   // tail tile: guarded stores
#pragma unroll
    for (int nt = 0; nt < 4; ++nt) {
      const int col = nt * 16 + m;
      const float bs = ldsBias[col];
#pragma unroll
      for (int r = 0; r < 8; ++r) {
        const int rr = row0 + r + (hi ? 8 : 0);
        if (rr < nrows) {
          float v = acc[nt][r] + bs;
          if (RELU) v = fmaxf(v, 0.0f);
          out[(size_t)rr * 64 + col] = v;
        }
      }
    }
  }
}

// ------------------------------- batchnorm ---------------------------------

__global__ __launch_bounds__(256) void k_bn_stats(
    const float* __restrict__ h, float* __restrict__ stats, int nrows)
{
  const int c = threadIdx.x & 63;
  const int t = blockIdx.x * 4 + (threadIdx.x >> 6);
  const int stride = gridDim.x * 4;
  float s = 0.f, q = 0.f;
  for (int r = t; r < nrows; r += stride) {
    float v = h[(size_t)r * 64 + c];
    s += v; q += v * v;
  }
  atomicAdd(&stats[c], s);
  atomicAdd(&stats[64 + c], q);
}

__global__ __launch_bounds__(256) void k_bn_apply(
    const float* __restrict__ h, const float* __restrict__ stats,
    const float* __restrict__ gamma, const float* __restrict__ beta,
    float* __restrict__ out, int nrows)
{
  const float invn = 1.0f / (float)nrows;
  size_t total = (size_t)nrows * 64;
  for (size_t idx = (size_t)blockIdx.x * 256 + threadIdx.x; idx < total;
       idx += (size_t)gridDim.x * 256) {
    int c = (int)(idx & 63);
    float mean = stats[c] * invn;
    float var  = stats[64 + c] * invn - mean * mean;
    float v = (h[idx] - mean) * rsqrtf(var + BN_EPS) * gamma[c] + beta[c];
    out[idx] = fmaxf(v, 0.0f);
  }
}

// -------------------------------- pooling ----------------------------------

__global__ __launch_bounds__(256) void k_pool(
    const float* __restrict__ h, const int* __restrict__ batch,
    float* __restrict__ sums, float* __restrict__ counts, int nrows)
{
  size_t total = (size_t)nrows * 64;
  for (size_t idx = (size_t)blockIdx.x * 256 + threadIdx.x; idx < total;
       idx += (size_t)gridDim.x * 256) {
    int row = (int)(idx >> 6), c = (int)(idx & 63);
    int g = batch[row];
    atomicAdd(&sums[(size_t)g * 64 + c], h[idx]);
    if (c == 0) atomicAdd(&counts[g], 1.0f);
  }
}

__global__ __launch_bounds__(256) void k_finalize(
    float* __restrict__ out, const float* __restrict__ counts, int ngraphs)
{
  int idx = blockIdx.x * 256 + threadIdx.x;
  if (idx < ngraphs * 64) out[idx] /= fmaxf(counts[idx >> 6], 1.0f);
}

// -------------------------------- launcher ---------------------------------

extern "C" void kernel_launch(void* const* d_in, const int* in_sizes, int n_in,
                              void* d_out, int out_size, void* d_ws, size_t ws_size,
                              hipStream_t stream) {
  const float* x      = (const float*)d_in[0];
  const int*   ei     = (const int*)d_in[1];   // (2,E): src=ei[0:E), dst=ei[E:2E)
  const int*   batch  = (const int*)d_in[2];
  const float* W1a    = (const float*)d_in[3];
  const float* b1a    = (const float*)d_in[4];
  const float* W2a    = (const float*)d_in[5];
  const float* b2a    = (const float*)d_in[6];
  const float* gamma1 = (const float*)d_in[7];
  const float* beta1  = (const float*)d_in[8];
  const float* W1b    = (const float*)d_in[9];
  const float* b1b    = (const float*)d_in[10];
  const float* W2b    = (const float*)d_in[11];
  const float* b2b    = (const float*)d_in[12];
  const float* gamma2 = (const float*)d_in[13];
  const float* beta2  = (const float*)d_in[14];

  const int N = in_sizes[0] / 128;     // 100000
  const int E = in_sizes[1] / 2;       // 1600000
  const int G = out_size / 64;         // 512

  // workspace layout (floats)
  float* ws     = (float*)d_ws;
  float* agg1   = ws;                                   // N*128
  float* bufA   = agg1 + (size_t)N * 128;               // N*64
  float* bufB   = bufA + (size_t)N * 64;                // N*64
  float* bufC   = bufB + (size_t)N * 64;                // N*64
  float* stats  = bufC + (size_t)N * 64;                // 128
  float* counts = stats + 128;                          // G

  float* outp = (float*)d_out;

  const int gemm_blocks = (N + 127) / 128;
  const int ew_blocks   = (E + 7) / 8;                  // 8 waves/block, 1 edge/wave

  // --- GINConv 1 ---
  hipMemsetAsync(agg1, 0, (size_t)N * 128 * sizeof(float), stream);
  k_agg128<<<ew_blocks, 256, 0, stream>>>(x, ei, ei + E, agg1, E);
  k_gemm<128, true,  true ><<<gemm_blocks, 256, 0, stream>>>(x, agg1, W1a, b1a, bufA, N);
  k_gemm<64,  false, false><<<gemm_blocks, 256, 0, stream>>>(bufA, nullptr, W2a, b2a, bufB, N);

  // --- BN1 + relu ---
  hipMemsetAsync(stats, 0, 128 * sizeof(float), stream);
  k_bn_stats<<<512, 256, 0, stream>>>(bufB, stats, N);
  k_bn_apply<<<1024, 256, 0, stream>>>(bufB, stats, gamma1, beta1, bufA, N);

  // --- GINConv 2 ---
  hipMemsetAsync(bufC, 0, (size_t)N * 64 * sizeof(float), stream);
  k_agg64<<<ew_blocks, 256, 0, stream>>>(bufA, ei, ei + E, bufC, E);
  k_gemm<64, true,  true ><<<gemm_blocks, 256, 0, stream>>>(bufA, bufC, W1b, b1b, bufB, N);
  k_gemm<64, false, false><<<gemm_blocks, 256, 0, stream>>>(bufB, nullptr, W2b, b2b, bufC, N);

  // --- BN2 + relu ---
  hipMemsetAsync(stats, 0, 128 * sizeof(float), stream);
  k_bn_stats<<<512, 256, 0, stream>>>(bufC, stats, N);
  k_bn_apply<<<1024, 256, 0, stream>>>(bufC, stats, gamma2, beta2, bufB, N);

  // --- global mean pool ---
  hipMemsetAsync(outp,   0, (size_t)G * 64 * sizeof(float), stream);
  hipMemsetAsync(counts, 0, (size_t)G * sizeof(float), stream);
  k_pool<<<1024, 256, 0, stream>>>(bufB, batch, outp, counts, N);
  k_finalize<<<(G * 64 + 255) / 256, 256, 0, stream>>>(outp, counts, G);
}
